// GINNet_20083267076738
// MI455X (gfx1250) — compile-verified, matrix operand order checked by reading, not run on hardware
//
#include <hip/hip_runtime.h>
#include <hip/hip_bf16.h>

// ---------- WMMA vector types (gfx1250, wave32) ----------
typedef __attribute__((ext_vector_type(16))) __bf16 v16bf;
typedef __attribute__((ext_vector_type(8)))  float  v8f;
typedef __attribute__((ext_vector_type(4)))  unsigned int uint4u;

struct U128x2 { uint4u lo, hi; };
union CvtAB { U128x2 u; v16bf v; };

__device__ __forceinline__ unsigned short f32_to_bf16_rne(float f) {
    unsigned int u = __builtin_bit_cast(unsigned int, f);
    u += 0x7FFFu + ((u >> 16) & 1u);
    return (unsigned short)(u >> 16);
}

// ---------- simple elementwise / setup kernels ----------
__global__ void k_zero(float* __restrict__ p, size_t n) {
    size_t t = (size_t)blockIdx.x * blockDim.x + threadIdx.x;
    if (t < n) p[t] = 0.0f;
}

// x0 = embed[node_ids], h = x0 (32 lanes per node, float4 each)
__global__ void k_gather_embed(const int* __restrict__ node_ids,
                               const float* __restrict__ embed,
                               float* __restrict__ x0, float* __restrict__ h,
                               int nNodes) {
    int t = blockIdx.x * blockDim.x + threadIdx.x;
    int node = t >> 5, lane = t & 31;
    if (node >= nNodes) return;
    int id = node_ids[node];
    float4 v = ((const float4*)(embed + (size_t)id * 128))[lane];
    ((float4*)(x0 + (size_t)node * 128))[lane] = v;
    ((float4*)(h  + (size_t)node * 128))[lane] = v;
}

// h[dst] += x0[src] over all edges (L2-resident atomics)
__global__ void k_edge_scatter(const int* __restrict__ src, const int* __restrict__ dst,
                               const float* __restrict__ x0, float* __restrict__ h,
                               int nEdges) {
    int t = blockIdx.x * blockDim.x + threadIdx.x;
    int e = t >> 5, lane = t & 31;
    if (e >= nEdges) return;
    int s = src[e], d = dst[e];
    float4 v = ((const float4*)(x0 + (size_t)s * 128))[lane];
    float* hd = h + (size_t)d * 128 + lane * 4;
    atomicAdd(hd + 0, v.x);
    atomicAdd(hd + 1, v.y);
    atomicAdd(hd + 2, v.z);
    atomicAdd(hd + 3, v.w);
}

__global__ void k_f32_to_bf16(const float* __restrict__ in,
                              unsigned short* __restrict__ out, size_t n) {
    size_t t = (size_t)blockIdx.x * blockDim.x + threadIdx.x;
    if (t < n) out[t] = f32_to_bf16_rne(in[t]);
}

// W [K][Nn] f32 -> Wt [Nn][K] bf16  (B operand wants column-major == Wt row-major)
__global__ void k_transpose_to_bf16(const float* __restrict__ W,
                                    unsigned short* __restrict__ Wt, int K, int Nn) {
    int t = blockIdx.x * blockDim.x + threadIdx.x;
    if (t >= K * Nn) return;
    int k = t / Nn, n = t - k * Nn;
    Wt[(size_t)n * K + k] = f32_to_bf16_rne(W[t]);
}

// ---------- WMMA GEMM with N register-blocking ----------
// C[M][Nn] = A[M][K](bf16) @ Bt[Nn][K](bf16)^T + bias
// One wave computes a 16 x (16*NT) strip: A fragment loaded once per K-step,
// reused across NT WMMAs (cuts A L2 traffic by NT).
template <int K, int NT>
__global__ void __launch_bounds__(256)
k_gemm_bf16(const unsigned short* __restrict__ A,
            const unsigned short* __restrict__ Bt,
            const float* __restrict__ bias,
            float* __restrict__ C, int M, int Nn) {
    int wave = blockIdx.x * (blockDim.x >> 5) + (threadIdx.x >> 5);
    int lane = threadIdx.x & 31;
    int groupsN = (Nn >> 4) / NT;
    int tileM = wave / groupsN;
    int gN = wave - tileM * groupsN;
    if (tileM * 16 >= M) return;                 // uniform per-wave exit
    int hf = lane >> 4;                          // which K-half this lane covers
    int r  = lane & 15;                          // A row / B column within tile
    int arow = tileM * 16 + r; if (arow >= M) arow = M - 1;   // clamp, keep EXEC full
    int n0 = gN * NT * 16;
    const unsigned short* Ap = A  + (size_t)arow * K + hf * 8;
    const unsigned short* Bp = Bt + (size_t)(n0 + r) * K + hf * 8;
    v8f acc[NT];
#pragma unroll
    for (int j = 0; j < NT; ++j) acc[j] = v8f{};
#pragma unroll
    for (int kb = 0; kb < K; kb += 32) {
        CvtAB a;
        a.u.lo = *(const uint4u*)(Ap + kb);       // K = kb + hf*8 + 0..7
        a.u.hi = *(const uint4u*)(Ap + kb + 16);  // K = kb + 16 + hf*8 + 0..7
#pragma unroll
        for (int j = 0; j < NT; ++j) {
            const unsigned short* bp = Bp + (size_t)j * 16 * K;  // folds to imm offset
            CvtAB b;
            b.u.lo = *(const uint4u*)(bp + kb);
            b.u.hi = *(const uint4u*)(bp + kb + 16);
            acc[j] = __builtin_amdgcn_wmma_f32_16x16x32_bf16(
                false, a.v, false, b.v, (short)0, acc[j], false, false);
        }
    }
#pragma unroll
    for (int j = 0; j < NT; ++j) {
        int n = n0 + j * 16 + r;
        float bv = bias[n];
#pragma unroll
        for (int v = 0; v < 8; ++v) {             // D layout: m = v + 8*hf
            int m = tileM * 16 + hf * 8 + v;
            if (m < M) C[(size_t)m * Nn + n] = acc[j][v] + bv;
        }
    }
}

// ---------- BatchNorm: column stats, finalize, apply+ReLU+bf16 ----------
__global__ void k_col_stats(const float* __restrict__ y, float* __restrict__ sums,
                            float* __restrict__ sq, int M, int Nn) {
    int c = threadIdx.x;                          // blockDim.x == Nn
    int chunk = (M + gridDim.x - 1) / gridDim.x;
    int r0 = blockIdx.x * chunk;
    int r1 = r0 + chunk; if (r1 > M) r1 = M;
    float s = 0.f, s2 = 0.f;
    for (int r = r0; r < r1; ++r) {
        float v = y[(size_t)r * Nn + c];
        s += v; s2 += v * v;
    }
    atomicAdd(&sums[c], s);
    atomicAdd(&sq[c], s2);
}

__global__ void k_bn_finalize(const float* __restrict__ sums, const float* __restrict__ sq,
                              const float* __restrict__ gamma, const float* __restrict__ beta,
                              float* __restrict__ scale, float* __restrict__ shift,
                              int Nn, float invM) {
    int c = blockIdx.x * blockDim.x + threadIdx.x;
    if (c >= Nn) return;
    float mu  = sums[c] * invM;
    float var = sq[c] * invM - mu * mu;
    float sc  = gamma[c] * rsqrtf(var + 1e-5f);
    scale[c] = sc;
    shift[c] = beta[c] - mu * sc;
}

__global__ void k_bn_relu_bf16(const float* __restrict__ y, const float* __restrict__ scale,
                               const float* __restrict__ shift, unsigned short* __restrict__ out,
                               size_t total) {
    size_t t = (size_t)blockIdx.x * blockDim.x + threadIdx.x;
    if (t >= total) return;
    int c = (int)(t & 255u);                      // Nn == 256
    float v = y[t] * scale[c] + shift[c];
    out[t] = f32_to_bf16_rne(v > 0.f ? v : 0.f);
}

// ---------- fused projection GEMMs + segment-sum scatter into d_out ----------
// One wave computes a full 16x128 output strip (NT=8): A fragments loaded once.
__global__ void __launch_bounds__(256)
k_gemm_score_scatter(const unsigned short* __restrict__ Xb,   // [M][128]
                     const unsigned short* __restrict__ P0t,  // [128][128]
                     const unsigned short* __restrict__ Hb,   // [M][256]
                     const unsigned short* __restrict__ P1t,  // [128][256]
                     const float* __restrict__ b0, const float* __restrict__ b1,
                     const int* __restrict__ batch,
                     float* __restrict__ out, int M) {
    const int NT = 8;                              // 8 * 16 = 128 = full output width
    int tileM = blockIdx.x * (blockDim.x >> 5) + (threadIdx.x >> 5);
    int lane = threadIdx.x & 31;
    if (tileM * 16 >= M) return;
    int hf = lane >> 4, r = lane & 15;
    int arow = tileM * 16 + r; if (arow >= M) arow = M - 1;
    v8f acc[NT];
#pragma unroll
    for (int j = 0; j < NT; ++j) acc[j] = v8f{};
    {   // score += x0 @ P0   (K = 128)
        const unsigned short* Ap = Xb  + (size_t)arow * 128 + hf * 8;
        const unsigned short* Bp = P0t + (size_t)r * 128 + hf * 8;
#pragma unroll
        for (int kb = 0; kb < 128; kb += 32) {
            CvtAB a;
            a.u.lo = *(const uint4u*)(Ap + kb);
            a.u.hi = *(const uint4u*)(Ap + kb + 16);
#pragma unroll
            for (int j = 0; j < NT; ++j) {
                const unsigned short* bp = Bp + (size_t)j * 16 * 128;
                CvtAB b;
                b.u.lo = *(const uint4u*)(bp + kb);
                b.u.hi = *(const uint4u*)(bp + kb + 16);
                acc[j] = __builtin_amdgcn_wmma_f32_16x16x32_bf16(
                    false, a.v, false, b.v, (short)0, acc[j], false, false);
            }
        }
    }
    {   // score += h1 @ P1   (K = 256)
        const unsigned short* Ap = Hb  + (size_t)arow * 256 + hf * 8;
        const unsigned short* Bp = P1t + (size_t)r * 256 + hf * 8;
#pragma unroll
        for (int kb = 0; kb < 256; kb += 32) {
            CvtAB a;
            a.u.lo = *(const uint4u*)(Ap + kb);
            a.u.hi = *(const uint4u*)(Ap + kb + 16);
#pragma unroll
            for (int j = 0; j < NT; ++j) {
                const unsigned short* bp = Bp + (size_t)j * 16 * 256;
                CvtAB b;
                b.u.lo = *(const uint4u*)(bp + kb);
                b.u.hi = *(const uint4u*)(bp + kb + 16);
                acc[j] = __builtin_amdgcn_wmma_f32_16x16x32_bf16(
                    false, a.v, false, b.v, (short)0, acc[j], false, false);
            }
        }
    }
#pragma unroll
    for (int v = 0; v < 8; ++v) {                  // D layout: m = v + 8*hf
        int m = tileM * 16 + hf * 8 + v;
        if (m < M) {
            int g = batch[m];
            float* og = out + (size_t)g * 128;
#pragma unroll
            for (int j = 0; j < NT; ++j) {
                int n = j * 16 + r;
                atomicAdd(&og[n], acc[j][v] + b0[n] + b1[n]);
            }
        }
    }
}

__global__ void k_counts(const int* __restrict__ batch, float* __restrict__ cnt, int n) {
    int t = blockIdx.x * blockDim.x + threadIdx.x;
    if (t < n) atomicAdd(&cnt[batch[t]], 1.0f);
}

__global__ void k_div(float* __restrict__ out, const float* __restrict__ cnt, int total) {
    int t = blockIdx.x * blockDim.x + threadIdx.x;
    if (t >= total) return;
    float c = cnt[t >> 7];
    out[t] /= (c > 1.0f ? c : 1.0f);
}

static inline int ceildiv(long long a, int b) { return (int)((a + b - 1) / b); }

extern "C" void kernel_launch(void* const* d_in, const int* in_sizes, int n_in,
                              void* d_out, int out_size, void* d_ws, size_t ws_size,
                              hipStream_t stream) {
    const int*   node_ids = (const int*)d_in[0];
    const int*   edge_idx = (const int*)d_in[1];
    const int*   batch    = (const int*)d_in[2];
    const float* embed    = (const float*)d_in[3];
    const float* W1  = (const float*)d_in[4];
    const float* b1  = (const float*)d_in[5];
    const float* g1  = (const float*)d_in[6];
    const float* be1 = (const float*)d_in[7];
    const float* W2  = (const float*)d_in[8];
    const float* b2  = (const float*)d_in[9];
    const float* g2  = (const float*)d_in[10];
    const float* be2 = (const float*)d_in[11];
    const float* P0w = (const float*)d_in[12];
    const float* P0b = (const float*)d_in[13];
    const float* P1w = (const float*)d_in[14];
    const float* P1b = (const float*)d_in[15];
    float* out = (float*)d_out;

    const int N = in_sizes[0];           // 100000 nodes
    const int E = in_sizes[1] / 2;       // 1.6M edges
    const int G = out_size / 128;        // 1000 graphs
    const int DIN = 128, DH = 256;
    const int* src = edge_idx;
    const int* dst = edge_idx + E;

    // workspace layout
    char* ws = (char*)d_ws;
    size_t off = 0;
    auto alloc = [&](size_t bytes) -> void* {
        void* p = ws + off;
        off += (bytes + 255) & ~(size_t)255;
        return p;
    };
    float* x0 = (float*)alloc((size_t)N * DIN * 4);
    float* h  = (float*)alloc((size_t)N * DIN * 4);
    float* y  = (float*)alloc((size_t)N * DH * 4);       // reused for GEMM1 and GEMM2 outputs
    unsigned short* x0b = (unsigned short*)alloc((size_t)N * DIN * 2);
    unsigned short* hb  = (unsigned short*)alloc((size_t)N * DIN * 2);
    unsigned short* zb  = (unsigned short*)alloc((size_t)N * DH * 2);  // z1, then h1
    unsigned short* W1t = (unsigned short*)alloc((size_t)DIN * DH * 2);
    unsigned short* W2t = (unsigned short*)alloc((size_t)DH * DH * 2);
    unsigned short* P0t = (unsigned short*)alloc((size_t)DIN * DIN * 2);
    unsigned short* P1t = (unsigned short*)alloc((size_t)DH * DIN * 2);
    float* stats = (float*)alloc((size_t)(8 * 256 + G) * 4);
    float *sumsA = stats, *sqA = stats + 256, *sumsB = stats + 512, *sqB = stats + 768;
    float *scaleA = stats + 1024, *shiftA = stats + 1280;
    float *scaleB = stats + 1536, *shiftB = stats + 1792;
    float *cnt = stats + 2048;

    const int T = 256;

    // zero accumulators (every call: deterministic)
    k_zero<<<ceildiv(out_size, T), T, 0, stream>>>(out, (size_t)out_size);
    k_zero<<<ceildiv(2048 + G, T), T, 0, stream>>>(stats, (size_t)(2048 + G));

    // gather + GIN aggregation
    k_gather_embed<<<ceildiv((long long)N * 32, T), T, 0, stream>>>(node_ids, embed, x0, h, N);
    k_edge_scatter<<<ceildiv((long long)E * 32, T), T, 0, stream>>>(src, dst, x0, h, E);

    // bf16 conversions of activations + weights (weights: transpose to [N][K])
    k_f32_to_bf16<<<ceildiv((long long)N * DIN, T), T, 0, stream>>>(x0, x0b, (size_t)N * DIN);
    k_f32_to_bf16<<<ceildiv((long long)N * DIN, T), T, 0, stream>>>(h, hb, (size_t)N * DIN);
    k_transpose_to_bf16<<<ceildiv(DIN * DH, T), T, 0, stream>>>(W1, W1t, DIN, DH);
    k_transpose_to_bf16<<<ceildiv(DH * DH, T), T, 0, stream>>>(W2, W2t, DH, DH);
    k_transpose_to_bf16<<<ceildiv(DIN * DIN, T), T, 0, stream>>>(P0w, P0t, DIN, DIN);
    k_transpose_to_bf16<<<ceildiv(DH * DIN, T), T, 0, stream>>>(P1w, P1t, DH, DIN);

    // GEMM1: y = h @ W1 + b1   (K=128, Nn=256, NT=4)
    int waves12 = ceildiv(N, 16) * ((DH / 16) / 4);
    k_gemm_bf16<128, 4><<<ceildiv(waves12, 8), 256, 0, stream>>>(hb, W1t, b1, y, N, DH);
    k_col_stats<<<128, DH, 0, stream>>>(y, sumsA, sqA, N, DH);
    k_bn_finalize<<<1, DH, 0, stream>>>(sumsA, sqA, g1, be1, scaleA, shiftA, DH, 1.0f / N);
    k_bn_relu_bf16<<<ceildiv((long long)N * DH, T), T, 0, stream>>>(y, scaleA, shiftA, zb, (size_t)N * DH);

    // GEMM2: y = z1 @ W2 + b2  (K=256, Nn=256, NT=4)
    k_gemm_bf16<256, 4><<<ceildiv(waves12, 8), 256, 0, stream>>>(zb, W2t, b2, y, N, DH);
    k_col_stats<<<128, DH, 0, stream>>>(y, sumsB, sqB, N, DH);
    k_bn_finalize<<<1, DH, 0, stream>>>(sumsB, sqB, g2, be2, scaleB, shiftB, DH, 1.0f / N);
    k_bn_relu_bf16<<<ceildiv((long long)N * DH, T), T, 0, stream>>>(y, scaleB, shiftB, zb, (size_t)N * DH);

    // fused projections + segment sum into d_out (one wave per 16-row tile)
    int waves3 = ceildiv(N, 16);
    k_gemm_score_scatter<<<ceildiv(waves3, 8), 256, 0, stream>>>(x0b, P0t, zb, P1t,
                                                                 P0b, P1b, batch, out, N);
    // mean
    k_counts<<<ceildiv(N, T), T, 0, stream>>>(batch, cnt, N);
    k_div<<<ceildiv(G * 128, T), T, 0, stream>>>(out, cnt, G * 128);
}
